// DualGNN_58188216926736
// MI455X (gfx1250) — matrix operand order repaired and checked by reading, not compiled, not run on hardware
//
#include <hip/hip_runtime.h>
#include <hip/hip_bf16.h>

typedef __attribute__((ext_vector_type(16))) _Float16 v16h;
typedef __attribute__((ext_vector_type(8)))  float    v8f;

#define IN_DIM  256
#define OUT_DIM 64
#define WPB     4          // waves per block in the WMMA MLP kernel

// ---------------------------------------------------------------------------
// WMMA fragment loaders (CDNA5 ISA 7.12.2 layouts, wave32)
// ---------------------------------------------------------------------------
// A (16x32 f16) from a row-major LDS tile: lanes 0-15 -> M=lane, K base 0;
// lanes 16-31 -> K bases 8 / 24. Both 8-half runs are contiguous -> 2x b128.
__device__ __forceinline__ v16h load_a_frag_lds(const _Float16* tile /*16 x IN_DIM*/,
                                                int lane, int k0) {
  const int m  = lane & 15;
  const int kb = (lane >> 4) * 8;
  const _Float16* rowp = tile + m * IN_DIM + k0 + kb;
  v16h a;
#pragma unroll
  for (int h = 0; h < 8; ++h) a[h] = rowp[h];
#pragma unroll
  for (int h = 0; h < 8; ++h) a[8 + h] = rowp[16 + h];
  return a;
}

// B fragment from the pre-packed f16 weight image: each lane reads its own
// contiguous 32 B (one v16h) -> two coalesced global_load_b128, no cvt.
__device__ __forceinline__ v16h load_b_frag_packed(const _Float16* __restrict__ P,
                                                   int ctTiles, int kt, int ct, int lane) {
  const _Float16* p = P + (((size_t)(kt * ctTiles + ct) * 32 + lane) << 4);
  return *(const v16h*)p;
}

// ---------------------------------------------------------------------------
// One-time weight pre-pack: W (K x Nc, f32 row-major) -> fragment-contiguous
// f16 image. Packed half index = ((kt*ctTiles + ct)*32 + lane)*16 + h,
// where k = kt*32 + (lane>>4)*8 + (h&7) + (h>>3)*16, n = ct*16 + (lane&15).
// ---------------------------------------------------------------------------
__global__ void pack_w_kernel(const float* __restrict__ W, _Float16* __restrict__ P,
                              int K, int Nc) {
  const int idx = blockIdx.x * blockDim.x + threadIdx.x;
  if (idx >= K * Nc) return;
  const int h    = idx & 15;
  const int l    = (idx >> 4) & 31;
  const int frag = idx >> 9;
  const int ctTiles = Nc >> 4;
  const int kt = frag / ctTiles, ct = frag - kt * ctTiles;
  const int k = kt * 32 + ((l >> 4) << 3) + (h & 7) + ((h >> 3) << 4);
  const int n = ct * 16 + (l & 15);
  P[idx] = (_Float16)W[(size_t)k * Nc + n];
}

// ---------------------------------------------------------------------------
// Fused 2-layer MLP:  Y = scale * (relu(X@W1+b1) @ W2 + b2), OUT_DIM = 64.
// One wave per 16-row tile; A fragments held in registers across column tiles.
// ---------------------------------------------------------------------------
__global__ __launch_bounds__(WPB * 32)
void mlp2_wmma_kernel(const float* __restrict__ X,
                      const _Float16* __restrict__ W1p, const float* __restrict__ B1,
                      const _Float16* __restrict__ W2p, const float* __restrict__ B2,
                      float* __restrict__ Y, int n, int hid, float scale) {
  __shared__ _Float16 Xs[WPB][16][IN_DIM];   // 32 KB
  __shared__ _Float16 Hs[WPB][16][IN_DIM];   // 32 KB (hid <= 256)

  const int wave = threadIdx.x >> 5;
  const int lane = threadIdx.x & 31;
  const int nTiles = (n + 15) >> 4;
  int tile = blockIdx.x * WPB + wave;
  if (tile >= nTiles) tile = nTiles - 1;      // clamp -> barriers stay uniform
  const int row0 = tile << 4;

  // stage X tile, fp32 -> fp16 (conversion out of the hot loop)
  for (int i = lane; i < 16 * (IN_DIM / 4); i += 32) {
    const int r = i >> 6;
    const int c = (i & 63) << 2;
    int rr = row0 + r; if (rr >= n) rr = n - 1;
    const float4 v = *(const float4*)(X + (size_t)rr * IN_DIM + c);
    Xs[wave][r][c + 0] = (_Float16)v.x;
    Xs[wave][r][c + 1] = (_Float16)v.y;
    Xs[wave][r][c + 2] = (_Float16)v.z;
    Xs[wave][r][c + 3] = (_Float16)v.w;
  }
  __syncthreads();

  // hoist all 8 A fragments (16x256 tile) into registers
  v16h afr[IN_DIM / 32];
#pragma unroll
  for (int kt = 0; kt < IN_DIM / 32; ++kt)
    afr[kt] = load_a_frag_lds(&Xs[wave][0][0], lane, kt * 32);

  // layer 1: H = relu(X @ W1 + b1), hid in {128, 256}
  const int hidTiles = hid >> 4;
#pragma unroll 1
  for (int ct = 0; ct < hidTiles; ++ct) {
    v8f acc = {};
#pragma unroll
    for (int kt = 0; kt < IN_DIM / 32; ++kt) {
      v16h b = load_b_frag_packed(W1p, hidTiles, kt, ct, lane);
      acc = __builtin_amdgcn_wmma_f32_16x16x32_f16(false, afr[kt], false, b,
                                                   (short)0, acc, false, false);
    }
    const int nn = ct * 16 + (lane & 15);
    const float bn = B1[nn];
#pragma unroll
    for (int v = 0; v < 8; ++v) {
      const int m = v + ((lane >> 4) << 3);
      const float hv = acc[v] + bn;
      Hs[wave][m][nn] = (_Float16)(hv > 0.f ? hv : 0.f);
    }
  }
  __syncthreads();

  // hoist H fragments (K = hid) into the same register array
  const int kTiles2 = hid >> 5;
#pragma unroll 1
  for (int kt = 0; kt < kTiles2; ++kt)
    afr[kt] = load_a_frag_lds(&Hs[wave][0][0], lane, kt * 32);

  // layer 2: Y = scale * (H @ W2 + b2)
#pragma unroll 1
  for (int ct = 0; ct < OUT_DIM / 16; ++ct) {
    v8f acc = {};
#pragma unroll 1
    for (int kt = 0; kt < kTiles2; ++kt) {
      v16h b = load_b_frag_packed(W2p, OUT_DIM / 16, kt, ct, lane);
      acc = __builtin_amdgcn_wmma_f32_16x16x32_f16(false, afr[kt], false, b,
                                                   (short)0, acc, false, false);
    }
    const int nn = ct * 16 + (lane & 15);
    const float bn = B2[nn];
#pragma unroll
    for (int v = 0; v < 8; ++v) {
      const int m = v + ((lane >> 4) << 3);
      int rr = row0 + m; if (rr >= n) rr = n - 1;
      Y[(size_t)rr * OUT_DIM + nn] = scale * (acc[v] + bn);
    }
  }
}

// ---------------------------------------------------------------------------
// gcn_norm pieces: deg = 1 (self loop) + sum_e w[e] at col[e];  dinv = rsqrt.
// ---------------------------------------------------------------------------
__global__ void init_deg_kernel(float* d1, float* d2, int n) {
  const int i = blockIdx.x * blockDim.x + threadIdx.x;
  if (i < n) { d1[i] = 1.0f; d2[i] = 1.0f; }
}

__device__ __forceinline__ void edge_weights(float w, float& w1, float& w2) {
  w1 = 1e-4f + 0.9999f * (1e-5f + w * 0.99998f);
  w2 = 1e-4f + 0.9999f * (0.99999f - w * 0.99998f);
}

__global__ void deg_kernel(const int* __restrict__ col, const float* __restrict__ ew,
                           float* deg1, float* deg2, int e) {
  const int i = blockIdx.x * blockDim.x + threadIdx.x;
  if (i >= e) return;
  float w1, w2; edge_weights(ew[i], w1, w2);
  const int c = col[i];
  atomicAdd(&deg1[c], w1);
  atomicAdd(&deg2[c], w2);
}

__global__ void rsqrt_kernel(float* d1, float* d2, int n) {
  const int i = blockIdx.x * blockDim.x + threadIdx.x;
  if (i >= n) return;
  const float a = d1[i]; d1[i] = (a > 0.f) ? rsqrtf(a) : 0.f;
  const float b = d2[i]; d2[i] = (b > 0.f) ? rsqrtf(b) : 0.f;
}

__global__ void norm_kernel(const int* __restrict__ row, const int* __restrict__ col,
                            const float* __restrict__ ew,
                            const float* __restrict__ dinv1, const float* __restrict__ dinv2,
                            float* __restrict__ nrm1, float* __restrict__ nrm2, int e) {
  const int i = blockIdx.x * blockDim.x + threadIdx.x;
  if (i >= e) return;
  float w1, w2; edge_weights(ew[i], w1, w2);
  const int r = row[i], c = col[i];
  nrm1[i] = dinv1[r] * w1 * dinv1[c];
  nrm2[i] = dinv2[r] * w2 * dinv2[c];
}

// ---------------------------------------------------------------------------
// Edge scatter: agg[col] += norm * src[row]   (8 threads/edge, float4 gathers,
// f32 atomics resolve in the 192 MB L2 since agg is only 25.6 MB). Prefetch
// the gather row a few thousand edges ahead (global_prefetch_b8).
// ---------------------------------------------------------------------------
#define SCATTER_LOOKAHEAD 2048
__global__ void scatter_kernel(const int* __restrict__ row, const int* __restrict__ col,
                               const float* __restrict__ nrm,
                               const float* __restrict__ src, float* __restrict__ agg,
                               int e) {
  const long long t = (long long)blockIdx.x * blockDim.x + threadIdx.x;
  const int ed = (int)(t >> 3);
  if (ed >= e) return;
  const int f = ((int)t & 7) << 3;
  const int ep = ed + SCATTER_LOOKAHEAD;
  if (ep < e)
    __builtin_prefetch(src + (size_t)row[ep] * OUT_DIM + f, 0, 0);
  const int r = row[ed], c = col[ed];
  const float w = nrm[ed];
  const float4* s = (const float4*)(src + (size_t)r * OUT_DIM + f);
  const float4 a = s[0], b = s[1];
  float* d = agg + (size_t)c * OUT_DIM + f;
  atomicAdd(d + 0, w * a.x); atomicAdd(d + 1, w * a.y);
  atomicAdd(d + 2, w * a.z); atomicAdd(d + 3, w * a.w);
  atomicAdd(d + 4, w * b.x); atomicAdd(d + 5, w * b.y);
  atomicAdd(d + 6, w * b.z); atomicAdd(d + 7, w * b.w);
}

// next = (1-alpha) * (agg + dinv^2 * cur) + alpha * h0   (self loop analytic)
__global__ void combine_kernel(const float* __restrict__ agg, const float* __restrict__ cur,
                               const float* __restrict__ h0, const float* __restrict__ dinv,
                               float* __restrict__ next, long long total) {
  const long long i = (long long)blockIdx.x * blockDim.x + threadIdx.x;
  if (i >= total) return;
  const int node = (int)(i >> 6);
  float sn = dinv[node]; sn *= sn;
  next[i] = 0.9f * (agg[i] + sn * cur[i]) + 0.1f * h0[i];
}

// base += 0.001 * mlp2(A): one wave per node, 2 outputs per lane (wave32)
__global__ void mlpA_kernel(const float* __restrict__ A,
                            const float* __restrict__ W1, const float* __restrict__ B1,
                            const float* __restrict__ W2, const float* __restrict__ B2,
                            float* __restrict__ base, int n) {
  const int wave = threadIdx.x >> 5, lane = threadIdx.x & 31;
  const int node = blockIdx.x * (blockDim.x >> 5) + wave;
  if (node >= n) return;
  const float a = A[node];
  float acc0 = B2[lane], acc1 = B2[lane + 32];
#pragma unroll 4
  for (int k = 0; k < 256; ++k) {
    const float hk = fmaxf(a * W1[k] + B1[k], 0.0f);
    acc0 += hk * W2[k * OUT_DIM + lane];
    acc1 += hk * W2[k * OUT_DIM + lane + 32];
  }
  base[(size_t)node * OUT_DIM + lane]      += 0.001f * acc0;
  base[(size_t)node * OUT_DIM + lane + 32] += 0.001f * acc1;
}

// out = log_softmax(x1 - x2 + out) over the 64 features; wave32 shfl reduction
__global__ void final_kernel(const float* __restrict__ x1, const float* __restrict__ x2,
                             float* __restrict__ out, int n) {
  const int wave = threadIdx.x >> 5, lane = threadIdx.x & 31;
  const int node = blockIdx.x * (blockDim.x >> 5) + wave;
  if (node >= n) return;
  const size_t b = (size_t)node * OUT_DIM;
  const float v0 = x1[b + lane]      - x2[b + lane]      + out[b + lane];
  const float v1 = x1[b + lane + 32] - x2[b + lane + 32] + out[b + lane + 32];
  float m = fmaxf(v0, v1);
#pragma unroll
  for (int s = 16; s; s >>= 1) m = fmaxf(m, __shfl_xor(m, s, 32));
  float sum = __expf(v0 - m) + __expf(v1 - m);
#pragma unroll
  for (int s = 16; s; s >>= 1) sum += __shfl_xor(sum, s, 32);
  const float lse = m + __logf(sum);
  out[b + lane]      = v0 - lse;
  out[b + lane + 32] = v1 - lse;
}

// ---------------------------------------------------------------------------
extern "C" void kernel_launch(void* const* d_in, const int* in_sizes, int n_in,
                              void* d_out, int out_size, void* d_ws, size_t ws_size,
                              hipStream_t stream) {
  (void)n_in; (void)out_size; (void)ws_size;
  const float* X    = (const float*)d_in[0];
  const int*   EI   = (const int*)  d_in[1];
  const float* EW   = (const float*)d_in[2];
  const float* A    = (const float*)d_in[3];
  const float* m1W1 = (const float*)d_in[4],  *m1b1 = (const float*)d_in[5];
  const float* m1W2 = (const float*)d_in[6],  *m1b2 = (const float*)d_in[7];
  const float* m2W1 = (const float*)d_in[8],  *m2b1 = (const float*)d_in[9];
  const float* m2W2 = (const float*)d_in[10], *m2b2 = (const float*)d_in[11];
  const float* oW1  = (const float*)d_in[12], *ob1  = (const float*)d_in[13];
  const float* oW2  = (const float*)d_in[14], *ob2  = (const float*)d_in[15];
  const float* AW1  = (const float*)d_in[16], *Ab1  = (const float*)d_in[17];
  const float* AW2  = (const float*)d_in[18], *Ab2  = (const float*)d_in[19];

  const int N = in_sizes[0] / IN_DIM;
  const int E = in_sizes[2];
  const int* row = EI;
  const int* col = EI + E;

  const size_t NF = (size_t)N * OUT_DIM;
  float* ws = (float*)d_ws;
  float* h1 = ws;           float* h2 = h1 + NF;
  float* P1 = h2 + NF;      float* Q1 = P1 + NF;
  float* P2 = Q1 + NF;      float* Q2 = P2 + NF;
  float* agg1 = Q2 + NF;    float* agg2 = agg1 + NF;
  float* dinv1 = agg2 + NF; float* dinv2 = dinv1 + ((N + 15) & ~15);
  float* nrm1 = dinv2 + ((N + 15) & ~15);
  float* nrm2 = nrm1 + ((E + 15) & ~15);
  // packed f16 weight images (32B-aligned: all prior offsets are x16 floats)
  _Float16* p_m1W1 = (_Float16*)(nrm2 + ((E + 15) & ~15));
  _Float16* p_m1W2 = p_m1W1 + 256 * 128;
  _Float16* p_m2W1 = p_m1W2 + 128 * 64;
  _Float16* p_m2W2 = p_m2W1 + 256 * 128;
  _Float16* p_oW1  = p_m2W2 + 128 * 64;
  _Float16* p_oW2  = p_oW1  + 256 * 256;
  float* base = (float*)d_out;   // reused as the non-propagated additive term

  // one-time weight pre-pack into WMMA B-fragment layout (f16)
  pack_w_kernel<<<(256 * 128 + 255) / 256, 256, 0, stream>>>(m1W1, p_m1W1, 256, 128);
  pack_w_kernel<<<(128 * 64  + 255) / 256, 256, 0, stream>>>(m1W2, p_m1W2, 128, 64);
  pack_w_kernel<<<(256 * 128 + 255) / 256, 256, 0, stream>>>(m2W1, p_m2W1, 256, 128);
  pack_w_kernel<<<(128 * 64  + 255) / 256, 256, 0, stream>>>(m2W2, p_m2W2, 128, 64);
  pack_w_kernel<<<(256 * 256 + 255) / 256, 256, 0, stream>>>(oW1,  p_oW1,  256, 256);
  pack_w_kernel<<<(256 * 64  + 255) / 256, 256, 0, stream>>>(oW2,  p_oW2,  256, 64);

  // dense MLPs (WMMA): h1, h2, and base = 0.001 * x_off
  const int nTiles = (N + 15) / 16;
  const dim3 mgrid((nTiles + WPB - 1) / WPB), mblock(WPB * 32);
  mlp2_wmma_kernel<<<mgrid, mblock, 0, stream>>>(X, p_m1W1, m1b1, p_m1W2, m1b2, h1, N, 128, 1.0f);
  mlp2_wmma_kernel<<<mgrid, mblock, 0, stream>>>(X, p_m2W1, m2b1, p_m2W2, m2b2, h2, N, 128, 1.0f);
  mlp2_wmma_kernel<<<mgrid, mblock, 0, stream>>>(X, p_oW1,  ob1,  p_oW2,  ob2,  base, N, 256, 0.001f);
  mlpA_kernel<<<(N + 7) / 8, 256, 0, stream>>>(A, AW1, Ab1, AW2, Ab2, base, N);

  // gcn_norm precompute
  init_deg_kernel<<<(N + 255) / 256, 256, 0, stream>>>(dinv1, dinv2, N);
  deg_kernel<<<(E + 255) / 256, 256, 0, stream>>>(col, EW, dinv1, dinv2, E);
  rsqrt_kernel<<<(N + 255) / 256, 256, 0, stream>>>(dinv1, dinv2, N);
  norm_kernel<<<(E + 255) / 256, 256, 0, stream>>>(row, col, EW, dinv1, dinv2, nrm1, nrm2, E);

  // 5-hop APPNP on both branches
  const long long total = (long long)NF;
  const int cgrid = (int)((total + 255) / 256);
  const int sgrid = (int)(((long long)E * 8 + 255) / 256);

  const float* cur1 = h1; const float* cur2 = h2;
  float* nxt1 = P1; float* nxt2 = P2;
  for (int hop = 0; hop < 5; ++hop) {
    hipMemsetAsync(agg1, 0, NF * sizeof(float), stream);
    hipMemsetAsync(agg2, 0, NF * sizeof(float), stream);
    scatter_kernel<<<sgrid, 256, 0, stream>>>(row, col, nrm1, cur1, agg1, E);
    scatter_kernel<<<sgrid, 256, 0, stream>>>(row, col, nrm2, cur2, agg2, E);
    combine_kernel<<<cgrid, 256, 0, stream>>>(agg1, cur1, h1, dinv1, nxt1, total);
    combine_kernel<<<cgrid, 256, 0, stream>>>(agg2, cur2, h2, dinv2, nxt2, total);
    cur1 = nxt1; cur2 = nxt2;
    nxt1 = (cur1 == P1) ? Q1 : P1;
    nxt2 = (cur2 == P2) ? Q2 : P2;
  }

  // out = log_softmax(x1 - x2 + base)
  final_kernel<<<(N + 7) / 8, 256, 0, stream>>>(cur1, cur2, base, N);
}
